// TriSynthesisBlock_34540126994977
// MI455X (gfx1250) — matrix operand order repaired and checked by reading, not compiled, not run on hardware
//
#include <hip/hip_runtime.h>

typedef __attribute__((ext_vector_type(16))) _Float16 v16h;
typedef __attribute__((ext_vector_type(8)))  _Float16 v8h;
typedef __attribute__((ext_vector_type(8)))  float    v8f;

#define SQRT2F 1.4142135623730951f

// padded activation geometry: [N][130][136][C], pixel (r,c) -> row r+1, col c+4
#define PROWS 130
#define PCOLS 136

union V16HU { v16h v; struct { v8h lo; v8h hi; } s; };

// ---------------------------------------------------------------------------
// Kernel 0: grid-stride zero fill (for padded-buffer borders)
// ---------------------------------------------------------------------------
__global__ void zero_kernel(uint4* __restrict__ p, long n16) {
  long i = (long)blockIdx.x * blockDim.x + threadIdx.x;
  uint4 z; z.x = 0; z.y = 0; z.z = 0; z.w = 0;
  for (long k = i; k < n16; k += (long)gridDim.x * blockDim.x) p[k] = z;
}

// ---------------------------------------------------------------------------
// Kernel 1: weight prep: wsq[o][c] = sum_taps w^2 ; fp16 repack [tap][O][C]
// ---------------------------------------------------------------------------
__global__ void prep_kernel(const float* __restrict__ w0, const float* __restrict__ w1,
                            float* __restrict__ wsq0, float* __restrict__ wsq1,
                            _Float16* __restrict__ w0h, _Float16* __restrict__ w1h) {
  int idx = blockIdx.x * 256 + threadIdx.x;
  if (idx < 256 * 512) {                       // layer0: O=256, C=512
    int o = idx >> 9, c = idx & 511;
    const float* p = w0 + ((size_t)o * 512 + c) * 9;
    float s = 0.f;
#pragma unroll
    for (int t = 0; t < 9; ++t) {
      float v = p[t];
      s += v * v;
      w0h[((size_t)t * 256 + o) * 512 + c] = (_Float16)v;
    }
    wsq0[idx] = s;
  } else if (idx < 256 * 512 + 256 * 256) {    // layer1: O=256, C=256
    int k = idx - 256 * 512;
    int o = k >> 8, c = k & 255;
    const float* p = w1 + ((size_t)o * 256 + c) * 9;
    float s = 0.f;
#pragma unroll
    for (int t = 0; t < 9; ++t) {
      float v = p[t];
      s += v * v;
      w1h[((size_t)t * 256 + o) * 256 + c] = (_Float16)v;
    }
    wsq1[k] = s;
  }
}

// ---------------------------------------------------------------------------
// Kernel 2: style affines + demodulation coefficients, all 3 branches
// grid (4 n, 3 branch), 256 threads
// ---------------------------------------------------------------------------
__global__ void styles_kernel(const float* __restrict__ ws1, const float* __restrict__ ws2,
                              const float* __restrict__ ws3,
                              const float* __restrict__ a0w, const float* __restrict__ a0b,
                              const float* __restrict__ a1w, const float* __restrict__ a1b,
                              const float* __restrict__ arw, const float* __restrict__ arb,
                              const float* __restrict__ wsq0, const float* __restrict__ wsq1,
                              float* __restrict__ styles0, float* __restrict__ styles1,
                              float* __restrict__ styles2, float* __restrict__ dc0,
                              float* __restrict__ dc1) {
  const int n = blockIdx.x, b = blockIdx.y;
  const float* wsb = (b == 0) ? ws1 : (b == 1) ? ws2 : ws3;
  const float* w0 = wsb + (n * 3 + 0) * 512;
  const float* w1 = wsb + (n * 3 + 1) * 512;
  const float* w2 = wsb + (n * 3 + 2) * 512;
  __shared__ float s0[512];
  __shared__ float s1[256];
  const float inv = 0.044194173824159216f;   // 1/sqrt(512)
  const int tid = threadIdx.x;
  for (int i = tid; i < 512; i += 256) {
    float acc = 0.f;
    for (int k = 0; k < 512; ++k) acc += w0[k] * a0w[(size_t)i * 512 + k];
    float v = acc * inv + a0b[i];
    s0[i] = v;
    styles0[((size_t)(b * 4 + n)) * 512 + i] = v;
  }
  {
    float acc = 0.f, acc2 = 0.f;
    for (int k = 0; k < 512; ++k) {
      acc  += w1[k] * a1w[(size_t)tid * 512 + k];
      acc2 += w2[k] * arw[(size_t)tid * 512 + k];
    }
    float v = acc * inv + a1b[tid];
    s1[tid] = v;
    styles1[((size_t)(b * 4 + n)) * 256 + tid] = v;
    styles2[((size_t)(b * 4 + n)) * 256 + tid] = (acc2 * inv + arb[tid]) * 0.0625f; // /sqrt(256)
  }
  __syncthreads();
  {
    int o = tid;
    float acc = 0.f;
    for (int i = 0; i < 512; ++i) { float s = s0[i]; acc += s * s * wsq0[(size_t)o * 512 + i]; }
    dc0[((size_t)(b * 4 + n)) * 256 + o] = rsqrtf(acc + 1e-8f);
    float acc1 = 0.f;
    for (int i = 0; i < 256; ++i) { float s = s1[i]; acc1 += s * s * wsq1[(size_t)o * 256 + i]; }
    dc1[((size_t)(b * 4 + n)) * 256 + o] = rsqrtf(acc1 + 1e-8f);
  }
}

// ---------------------------------------------------------------------------
// Separable 2x upsample taps: out[p] = w0*in[i0] + w1*in[i0+1]
// ---------------------------------------------------------------------------
__device__ __forceinline__ void up_taps(int p, int& i0, float& w0, float& w1) {
  i0 = (p >> 1) + (p & 1) - 1;
  w0 = (p & 1) ? 0.75f : 0.25f;
  w1 = 1.f - w0;
}

// ---------------------------------------------------------------------------
// Kernel 3: modulate + upsample 64->128, write padded-NHWC fp16 (one branch)
// ---------------------------------------------------------------------------
__global__ void mod_upsample_kernel(const float* __restrict__ x, const float* __restrict__ styles0,
                                    _Float16* __restrict__ up) {
  int idx = blockIdx.x * 256 + threadIdx.x;     // 4*128*128*512 = 2^25
  int c  = idx & 511;
  int pc = (idx >> 9) & 127;
  int pr = (idx >> 16) & 127;
  int n  = idx >> 23;
  int ir0, ic0; float wr0, wr1, wc0, wc1;
  up_taps(pr, ir0, wr0, wr1);
  up_taps(pc, ic0, wc0, wc1);
  const float* src = x + ((size_t)(n * 512 + c)) * 64 * 64;
  float v = 0.f;
#pragma unroll
  for (int a = 0; a < 2; ++a) {
    int ir = ir0 + a; if ((unsigned)ir >= 64u) continue;
    float wr = a ? wr1 : wr0;
#pragma unroll
    for (int bb = 0; bb < 2; ++bb) {
      int ic = ic0 + bb; if ((unsigned)ic >= 64u) continue;
      v += wr * (bb ? wc1 : wc0) * src[ir * 64 + ic];
    }
  }
  size_t o = ((size_t)((n * PROWS + pr + 1) * PCOLS) + pc + 4) * 512 + c;
  up[o] = (_Float16)(v * styles0[(size_t)n * 512 + c]);
}

// ---------------------------------------------------------------------------
// Kernel 4: LDS-free implicit-GEMM 3x3 conv via V_WMMA_F32_16X16X32_F16.
// Software-pipelined: next K-chunk fragments prefetched while current WMMAs run.
// Block 256 thr = 8 waves; tile 128 oc x 64 px; wave = 32 oc x 32 px
// (2 A-frags x 2 B-frags -> 4 WMMA per K-step, 8 b128 loads per K-step).
// grid: x = row*2+colblock (256), y = mtile (2), z = n (4)
// ---------------------------------------------------------------------------
template <int IC, bool LAYER0>
__global__ __launch_bounds__(256) void conv3x3_wmma_kernel(
    const _Float16* __restrict__ inh,      // [N][130][136][IC]
    const _Float16* __restrict__ wh,       // [9][256][IC]
    const float* __restrict__ dcoefs,      // [N][256] (this branch)
    const float* __restrict__ bias,        // [256]
    const float* __restrict__ noise,       // [128*128]
    const float* __restrict__ ns,          // scalar
    const float* __restrict__ styles_next, // [N][256] (LAYER0 only)
    _Float16* __restrict__ out_h,          // LAYER0: padded NHWC fp16
    float* __restrict__ out_f) {           // layer1: NCHW f32 (d_out x slot)
  const int tid  = threadIdx.x;
  const int wave = tid >> 5, lane = tid & 31;
  const int nl   = lane & 15;
  const int hi   = lane >> 4;              // lane-group select
  const int ocg  = wave & 3;               // 4 oc strips of 32
  const int pxg  = wave >> 2;              // 2 px strips of 32
  const int r    = blockIdx.x >> 1;
  const int cb   = (blockIdx.x & 1) * 64;
  const int n    = blockIdx.z;
  const int oc_base = blockIdx.y * 128 + ocg * 32;
  const int pxbase  = cb + pxg * 32;

  v8f acc[2][2];
#pragma unroll
  for (int i = 0; i < 2; ++i)
#pragma unroll
    for (int j = 0; j < 2; ++j)
#pragma unroll
      for (int e = 0; e < 8; ++e) acc[i][j][e] = 0.f;

  const int kb2 = hi * 8;    // A: K-offset of lane group within each half
  const int kbB = hi * 8;    // B: K-offset of lane group

  for (int t = 0; t < 9; ++t) {
    const int kh = t / 3, kw = t % 3;
    const _Float16* aw0 = wh + ((size_t)t * 256 + oc_base + nl) * IC;
    const _Float16* aw1 = aw0 + (size_t)16 * IC;
    const _Float16* bp0 =
        inh + ((size_t)((n * PROWS + r + kh) * PCOLS) + (pxbase + nl + kw + 3)) * IC + kbB;
    const _Float16* bp1 = bp0 + (size_t)16 * IC;

    // prologue: fragments for c0 = 0
    V16HU a0, a1, b0, b1;
    a0.s.lo = *(const v8h*)(aw0 + kb2);      a0.s.hi = *(const v8h*)(aw0 + kb2 + 16);
    a1.s.lo = *(const v8h*)(aw1 + kb2);      a1.s.hi = *(const v8h*)(aw1 + kb2 + 16);
    b0.s.lo = *(const v8h*)(bp0);            b0.s.hi = *(const v8h*)(bp0 + 16);
    b1.s.lo = *(const v8h*)(bp1);            b1.s.hi = *(const v8h*)(bp1 + 16);

#pragma unroll 2
    for (int c0 = 0; c0 < IC; c0 += 32) {
      // prefetch c0+32 before consuming current fragments (unconditional;
      // one-chunk overfetch lands in the adjacent workspace carve - harmless)
      const int cn = c0 + 32;
      V16HU an0, an1, bn0, bn1;
      an0.s.lo = *(const v8h*)(aw0 + cn + kb2);  an0.s.hi = *(const v8h*)(aw0 + cn + kb2 + 16);
      an1.s.lo = *(const v8h*)(aw1 + cn + kb2);  an1.s.hi = *(const v8h*)(aw1 + cn + kb2 + 16);
      bn0.s.lo = *(const v8h*)(bp0 + cn);        bn0.s.hi = *(const v8h*)(bp0 + cn + 16);
      bn1.s.lo = *(const v8h*)(bp1 + cn);        bn1.s.hi = *(const v8h*)(bp1 + cn + 16);

      acc[0][0] = __builtin_amdgcn_wmma_f32_16x16x32_f16(
          false, a0.v, false, b0.v, (short)0, acc[0][0], false, false);
      acc[0][1] = __builtin_amdgcn_wmma_f32_16x16x32_f16(
          false, a0.v, false, b1.v, (short)0, acc[0][1], false, false);
      acc[1][0] = __builtin_amdgcn_wmma_f32_16x16x32_f16(
          false, a1.v, false, b0.v, (short)0, acc[1][0], false, false);
      acc[1][1] = __builtin_amdgcn_wmma_f32_16x16x32_f16(
          false, a1.v, false, b1.v, (short)0, acc[1][1], false, false);

      a0 = an0; a1 = an1; b0 = bn0; b1 = bn1;
    }
  }

  // ---- fused epilogue: demod + noise + bias + leakyrelu*sqrt2 ----
  const float nsv = ns[0];
#pragma unroll
  for (int i = 0; i < 2; ++i) {
    const int oc0 = oc_base + i * 16 + hi * 8;   // C-layout: M = hi*8 + vgpr
#pragma unroll
    for (int j = 0; j < 2; ++j) {
      const int p = pxbase + j * 16 + nl;        // C-layout: N = lane%16
      const float nz = noise[r * 128 + p] * nsv;
      if constexpr (LAYER0) {
        v8h oh;
#pragma unroll
        for (int v = 0; v < 8; ++v) {
          const int oc = oc0 + v;
          float val = acc[i][j][v] * dcoefs[(size_t)n * 256 + oc] + nz + bias[oc];
          val = (val < 0.f ? 0.2f * val : val) * SQRT2F;
          oh[v] = (_Float16)(val * styles_next[(size_t)n * 256 + oc]);
        }
        *(v8h*)(out_h + ((size_t)((n * PROWS + r + 1) * PCOLS) + p + 4) * 256 + oc0) = oh;
      } else {
#pragma unroll
        for (int v = 0; v < 8; ++v) {
          const int oc = oc0 + v;
          float val = acc[i][j][v] * dcoefs[(size_t)n * 256 + oc] + nz + bias[oc];
          val = (val < 0.f ? 0.2f * val : val) * SQRT2F;
          out_f[((size_t)(n * 256 + oc)) * 16384 + r * 128 + p] = val;
        }
      }
    }
  }
}

// ---------------------------------------------------------------------------
// Kernel 5: upsample the skip image 64->128 (one branch), f32 to d_out
// ---------------------------------------------------------------------------
__global__ void img_up_kernel(const float* __restrict__ img_in, float* __restrict__ img_out) {
  int idx = blockIdx.x * 256 + threadIdx.x;    // 4*3*128*128 = 196608
  if (idx >= 4 * 3 * 128 * 128) return;
  int pc = idx & 127, pr = (idx >> 7) & 127, nc = idx >> 14;  // nc = n*3+c
  int ir0, ic0; float wr0, wr1, wc0, wc1;
  up_taps(pr, ir0, wr0, wr1);
  up_taps(pc, ic0, wc0, wc1);
  const float* src = img_in + (size_t)nc * 64 * 64;
  float v = 0.f;
#pragma unroll
  for (int a = 0; a < 2; ++a) {
    int ir = ir0 + a; if ((unsigned)ir >= 64u) continue;
    float wr = a ? wr1 : wr0;
#pragma unroll
    for (int bb = 0; bb < 2; ++bb) {
      int ic = ic0 + bb; if ((unsigned)ic >= 64u) continue;
      v += wr * (bb ? wc1 : wc0) * src[ir * 64 + ic];
    }
  }
  img_out[idx] = v;
}

// ---------------------------------------------------------------------------
// Kernel 6: toRGB 1x1 conv (256 -> 3), accumulate into upsampled image
// grid (64 pixel-chunks, 3 o, 4 n)
// ---------------------------------------------------------------------------
__global__ void torgb_kernel(const float* __restrict__ xout, const float* __restrict__ styles2,
                             const float* __restrict__ rgbw, const float* __restrict__ rgbb,
                             float* __restrict__ img) {
  const int p = blockIdx.x * 256 + threadIdx.x;   // 0..16383
  const int o = blockIdx.y, n = blockIdx.z;
  __shared__ float sw[256];
  sw[threadIdx.x] = styles2[(size_t)n * 256 + threadIdx.x] * rgbw[(size_t)o * 256 + threadIdx.x];
  __syncthreads();
  float acc = rgbb[o];
  const float* xp = xout + (size_t)n * 256 * 16384 + p;
  for (int i = 0; i < 256; ++i) acc += xp[(size_t)i * 16384] * sw[i];
  size_t oi = ((size_t)n * 3 + o) * 16384 + p;
  img[oi] += acc;
}

// ---------------------------------------------------------------------------
extern "C" void kernel_launch(void* const* d_in, const int* in_sizes, int n_in,
                              void* d_out, int out_size, void* d_ws, size_t ws_size,
                              hipStream_t stream) {
  (void)in_sizes; (void)n_in; (void)out_size; (void)ws_size;
  const float* xin[3]   = {(const float*)d_in[0], (const float*)d_in[1], (const float*)d_in[2]};
  const float* imgin[3] = {(const float*)d_in[3], (const float*)d_in[4], (const float*)d_in[5]};
  const float* wsarr[3] = {(const float*)d_in[6], (const float*)d_in[7], (const float*)d_in[8]};
  const float* a0w = (const float*)d_in[9];
  const float* a0b = (const float*)d_in[10];
  const float* w0  = (const float*)d_in[11];
  const float* b0  = (const float*)d_in[12];
  const float* nc0 = (const float*)d_in[13];
  const float* ns0 = (const float*)d_in[14];
  const float* a1w = (const float*)d_in[15];
  const float* a1b = (const float*)d_in[16];
  const float* w1  = (const float*)d_in[17];
  const float* b1  = (const float*)d_in[18];
  const float* nc1 = (const float*)d_in[19];
  const float* ns1 = (const float*)d_in[20];
  const float* arw = (const float*)d_in[21];
  const float* arb = (const float*)d_in[22];
  const float* rgw = (const float*)d_in[23];
  const float* rgb = (const float*)d_in[24];

  // workspace carve-out
  char* wsp = (char*)d_ws;
  size_t off = 0;
  auto carve = [&](size_t bytes) -> void* {
    void* p = wsp + off;
    off = (off + bytes + 255) & ~(size_t)255;
    return p;
  };
  float*    styles0 = (float*)carve(3 * 4 * 512 * sizeof(float));
  float*    styles1 = (float*)carve(3 * 4 * 256 * sizeof(float));
  float*    styles2 = (float*)carve(3 * 4 * 256 * sizeof(float));
  float*    dc0     = (float*)carve(3 * 4 * 256 * sizeof(float));
  float*    dc1     = (float*)carve(3 * 4 * 256 * sizeof(float));
  float*    wsq0    = (float*)carve(256 * 512 * sizeof(float));
  float*    wsq1    = (float*)carve(256 * 256 * sizeof(float));
  _Float16* w0h     = (_Float16*)carve((size_t)9 * 256 * 512 * sizeof(_Float16));
  _Float16* w1h     = (_Float16*)carve((size_t)9 * 256 * 256 * sizeof(_Float16));
  const size_t up0_elems  = (size_t)4 * PROWS * PCOLS * 512;   // padded NHWC
  const size_t act0_elems = (size_t)4 * PROWS * PCOLS * 256;
  _Float16* up0   = (_Float16*)carve(up0_elems * sizeof(_Float16));
  _Float16* act0h = (_Float16*)carve(act0_elems * sizeof(_Float16));

  // zero padded buffers once per call (borders stay zero; interiors rewritten)
  zero_kernel<<<2048, 256, 0, stream>>>((uint4*)up0, (long)(up0_elems * 2 / 16));
  zero_kernel<<<2048, 256, 0, stream>>>((uint4*)act0h, (long)(act0_elems * 2 / 16));

  prep_kernel<<<768, 256, 0, stream>>>(w0, w1, wsq0, wsq1, w0h, w1h);
  styles_kernel<<<dim3(4, 3), 256, 0, stream>>>(wsarr[0], wsarr[1], wsarr[2],
      a0w, a0b, a1w, a1b, arw, arb, wsq0, wsq1, styles0, styles1, styles2, dc0, dc1);

  float* outf = (float*)d_out;
  for (int b = 0; b < 3; ++b) {
    float* xout_b = outf + (size_t)b * 4 * 256 * 128 * 128;
    float* img_b  = outf + (size_t)3 * 4 * 256 * 128 * 128 + (size_t)b * 4 * 3 * 128 * 128;

    mod_upsample_kernel<<<(4 * 512 * 128 * 128) / 256, 256, 0, stream>>>(
        xin[b], styles0 + (size_t)b * 4 * 512, up0);

    conv3x3_wmma_kernel<512, true><<<dim3(256, 2, 4), 256, 0, stream>>>(
        up0, w0h, dc0 + (size_t)b * 4 * 256, b0, nc0, ns0,
        styles1 + (size_t)b * 4 * 256, act0h, nullptr);

    conv3x3_wmma_kernel<256, false><<<dim3(256, 2, 4), 256, 0, stream>>>(
        act0h, w1h, dc1 + (size_t)b * 4 * 256, b1, nc1, ns1,
        nullptr, nullptr, xout_b);

    img_up_kernel<<<768, 256, 0, stream>>>(imgin[b], img_b);

    torgb_kernel<<<dim3(64, 3, 4), 256, 0, stream>>>(
        xout_b, styles2 + (size_t)b * 4 * 256, rgw, rgb, img_b);
  }
}
